// EdgeAwareModule_68281390072570
// MI455X (gfx1250) — matrix operand order repaired and checked by reading, not compiled, not run on hardware
//
#include <hip/hip_runtime.h>
#include <hip/hip_bf16.h>
#include <math.h>

typedef _Float16 h16t;
typedef __attribute__((ext_vector_type(16))) _Float16 v16h;
typedef __attribute__((ext_vector_type(8)))  _Float16 v8h;
typedef __attribute__((ext_vector_type(8)))  float    v8f;

#define BATCH 8
#define NPTS  4096
#define CH    256
#define BN    (BATCH * NPTS)
#define KNN   16
#define PROB_OFF ((size_t)BN * CH)

// ---------------- workspace layout (bytes) ----------------
// featF16 : BN*256 halfs            @ 0          (16,777,216 B)
// geoF16  : BN*16 halfs             @ 16777216   ( 1,048,576 B)
// rwT     : 256*256 halfs (N-major) @ 17825792   (   131,072 B)
// ew1T    : 64*288 halfs  (N-major) @ 17956864   (    36,864 B)
#define WS_FEAT 0
#define WS_GEO  16777216
#define WS_RWT  17825792
#define WS_EW1T 17956864

// ============================================================
// Prep kernels: f32 -> f16 conversions and weight transposes
// ============================================================
__global__ void cvt_feat_kernel(const float* __restrict__ f, h16t* __restrict__ o) {
    int i = blockIdx.x * 256 + threadIdx.x;
    o[i] = (h16t)f[i];
}

__global__ void tr_rw_kernel(const float* __restrict__ w, h16t* __restrict__ o) {
    // r_w is [K=256][N=256]; produce N-major: o[n*256 + k]
    int n = blockIdx.x, k = threadIdx.x;
    o[n * 256 + k] = (h16t)w[k * 256 + n];
}

__global__ void tr_ew1_kernel(const float* __restrict__ w, h16t* __restrict__ o) {
    // e_w1 is [K=272][N=64]; produce N-major padded to K=288: o[n*288 + k]
    int n = blockIdx.x, k = threadIdx.x;   // blockDim.x == 288
    float v = (k < 272) ? w[k * 64 + n] : 0.0f;
    o[n * 288 + k] = (h16t)v;
}

// ============================================================
// KNN (top-16 incl. self) + geometric features + geo MLP
// one thread per point; candidates tiled through LDS
// ============================================================
__global__ void knn_geo_kernel(const float* __restrict__ xyz,
                               const float* __restrict__ g_w1, const float* __restrict__ g_b1,
                               const float* __restrict__ g_lnw, const float* __restrict__ g_lnb,
                               const float* __restrict__ g_w2, const float* __restrict__ g_b2,
                               h16t* __restrict__ geoOut) {
    __shared__ float sx[256], sy[256], sz[256];
    const int tid = threadIdx.x;
    const int b = blockIdx.x >> 4;                 // 16 blocks per batch
    const int p = ((blockIdx.x & 15) << 8) + tid;  // point index in batch
    const float* base = xyz + (size_t)b * NPTS * 3;

    const float qx = base[p * 3 + 0];
    const float qy = base[p * 3 + 1];
    const float qz = base[p * 3 + 2];

    // sorted descending: bd[0] = current worst (largest) of the best-16
    float bd[KNN]; int bi[KNN];
#pragma unroll
    for (int j = 0; j < KNN; ++j) { bd[j] = 3.0e38f; bi[j] = 0; }

    for (int t0 = 0; t0 < NPTS; t0 += 256) {
        __syncthreads();
        int c = t0 + tid;
        sx[tid] = base[c * 3 + 0];
        sy[tid] = base[c * 3 + 1];
        sz[tid] = base[c * 3 + 2];
        __syncthreads();
        for (int j = 0; j < 256; ++j) {
            float dx = qx - sx[j], dy = qy - sy[j], dz = qz - sz[j];
            float d2 = dx * dx + dy * dy + dz * dz;
            if (d2 < bd[0]) {
                bd[0] = d2; bi[0] = t0 + j;
#pragma unroll
                for (int s = 0; s < KNN - 1; ++s) {
                    if (bd[s] < bd[s + 1]) {
                        float tf = bd[s]; bd[s] = bd[s + 1]; bd[s + 1] = tf;
                        int ti = bi[s]; bi[s] = bi[s + 1]; bi[s + 1] = ti;
                    }
                }
            }
        }
    }

    // gather neighbor coords, compute mean
    float nx[KNN], ny[KNN], nz[KNN];
    float mx = 0.f, my = 0.f, mz = 0.f;
#pragma unroll
    for (int j = 0; j < KNN; ++j) {
        int id = bi[j];
        nx[j] = base[id * 3 + 0]; ny[j] = base[id * 3 + 1]; nz[j] = base[id * 3 + 2];
        mx += nx[j]; my += ny[j]; mz += nz[j];
    }
    mx *= (1.f / KNN); my *= (1.f / KNN); mz *= (1.f / KNN);

    // centered stats: only diagonal covariance is needed
    float sxx = 0.f, syy = 0.f, szz = 0.f, dsum = 0.f, d2sum = 0.f;
#pragma unroll
    for (int j = 0; j < KNN; ++j) {
        float cx = nx[j] - mx, cy = ny[j] - my, cz = nz[j] - mz;
        float r2 = cx * cx + cy * cy + cz * cz;
        sxx += cx * cx; syy += cy * cy; szz += cz * cz;
        float dd = sqrtf(r2);
        dsum += dd; d2sum += r2;
    }
    float c00 = sxx * (1.f / KNN), c11 = syy * (1.f / KNN), c22 = szz * (1.f / KNN);
    float trace = c00 + c11 + c22;
    float det = fmaxf(c00 * c11 * c22, 1e-8f);
    float s0 = sqrtf(sxx * (1.f / 15.f));
    float s1 = sqrtf(syy * (1.f / 15.f));
    float s2 = sqrtf(szz * (1.f / 15.f));
    float md = dsum * (1.f / KNN);
    float vd = (d2sum - (float)KNN * md * md) * (1.f / 15.f);
    float sd = sqrtf(fmaxf(vd, 0.f));
    float t = trace + 1e-6f;

    float geo[10] = { trace, det, s0, s1, s2, md, sd, s0 / t, s1 / t, s2 / t };

    // geo MLP: Linear(10,32) -> LN(32) -> ReLU -> Linear(32,16)
    float h[32];
#pragma unroll
    for (int o = 0; o < 32; ++o) {
        float a = g_b1[o];
#pragma unroll
        for (int i = 0; i < 10; ++i) a += geo[i] * g_w1[i * 32 + o];
        h[o] = a;
    }
    float m = 0.f;
#pragma unroll
    for (int o = 0; o < 32; ++o) m += h[o];
    m *= (1.f / 32.f);
    float v = 0.f;
#pragma unroll
    for (int o = 0; o < 32; ++o) { float d = h[o] - m; v += d * d; }
    v *= (1.f / 32.f);
    float rs = rsqrtf(v + 1e-5f);
#pragma unroll
    for (int o = 0; o < 32; ++o)
        h[o] = fmaxf((h[o] - m) * rs * g_lnw[o] + g_lnb[o], 0.f);

    size_t bn = (size_t)b * NPTS + p;
#pragma unroll
    for (int o2 = 0; o2 < 16; ++o2) {
        float a = g_b2[o2];
#pragma unroll
        for (int o = 0; o < 32; ++o) a += h[o] * g_w2[o * 16 + o2];
        geoOut[bn * 16 + o2] = (h16t)a;
    }
}

// ============================================================
// Fused edge_pred + edge_refine via WMMA f32_16x16x32_f16
// 1 wave per 16-row tile; 4 waves / block; 64KB dynamic LDS
// ============================================================
__global__ void fused_edge_kernel(
    const h16t* __restrict__ featH, const h16t* __restrict__ geoH,
    const h16t* __restrict__ rwT, const h16t* __restrict__ ew1T,
    const float* __restrict__ featF,
    const float* __restrict__ e_b1, const float* __restrict__ e_lnw, const float* __restrict__ e_lnb,
    const float* __restrict__ e_w2, const float* __restrict__ e_b2,
    const float* __restrict__ r_b, const float* __restrict__ r_lnw, const float* __restrict__ r_lnb,
    float* __restrict__ out) {
    extern __shared__ float smem[];
    const int lane = threadIdx.x & 31;
    const int wave = threadIdx.x >> 5;
    const int m0 = (blockIdx.x * 4 + wave) * 16;

    float* refBuf = smem + wave * (16 * CH);  // 16x256 f32; edge phase uses first 16x64
    float* edgeBuf = refBuf;                  // aliased (edge fully consumed before refine store)

    const int arow = lane & 15;
    const int aoff = (lane < 16) ? 0 : 8;     // A: lanes<16 hold K{0..7,16..23}, else K{8..15,24..31}
    const int boff = (lane < 16) ? 0 : 16;    // B: unused; B loads use aoff, so B's K order matches A
    const int rbase = (lane < 16) ? 0 : 8;    // C/D: rows 0..7 vs 8..15
    const int colL = lane & 15;

    // ---- build A fragments: 8 feature k-tiles + 1 geo k-tile (zero-padded) ----
    v16h a[9];
    {
        const h16t* fp = featH + (size_t)(m0 + arow) * CH;
#pragma unroll
        for (int kt = 0; kt < 8; ++kt) {
            v8h lo = *reinterpret_cast<const v8h*>(fp + kt * 32 + aoff);
            v8h hi = *reinterpret_cast<const v8h*>(fp + kt * 32 + 16 + aoff);
            a[kt] = __builtin_shufflevector(lo, hi, 0, 1, 2, 3, 4, 5, 6, 7,
                                            8, 9, 10, 11, 12, 13, 14, 15);
        }
        v8h glo = *reinterpret_cast<const v8h*>(geoH + (size_t)(m0 + arow) * 16 + aoff);
        v8h z;
#pragma unroll
        for (int i = 0; i < 8; ++i) z[i] = (_Float16)0.0f;
        a[8] = __builtin_shufflevector(glo, z, 0, 1, 2, 3, 4, 5, 6, 7,
                                       8, 9, 10, 11, 12, 13, 14, 15);
    }

    // ---- edge_pred GEMM: [16x288] x [288x64] ----
#pragma unroll
    for (int nt = 0; nt < 4; ++nt) {
        v8f acc = {};
        const h16t* bp = ew1T + (size_t)(nt * 16 + colL) * 288 + aoff;
#pragma unroll
        for (int kt = 0; kt < 9; ++kt) {
            v8h blo = *reinterpret_cast<const v8h*>(bp + kt * 32);
            v8h bhi = *reinterpret_cast<const v8h*>(bp + kt * 32 + 16);
            v16h bb = __builtin_shufflevector(blo, bhi, 0, 1, 2, 3, 4, 5, 6, 7,
                                              8, 9, 10, 11, 12, 13, 14, 15);
            acc = __builtin_amdgcn_wmma_f32_16x16x32_f16(
                false, a[kt], false, bb, (short)0, acc, false, false);
        }
        int colD = nt * 16 + colL;
#pragma unroll
        for (int r = 0; r < 8; ++r) edgeBuf[(rbase + r) * 64 + colD] = acc[r];
    }
    __syncthreads();

    // ---- edge epilogue: +b, LN(64), ReLU, dot e_w2, sigmoid ----
    float p = 0.f;
    if (lane < 16) {
        float* hb = edgeBuf + lane * 64;
        float s = 0.f, s2 = 0.f;
        for (int c = 0; c < 64; ++c) {
            float x = hb[c] + e_b1[c];
            hb[c] = x; s += x; s2 += x * x;
        }
        float m = s * (1.f / 64.f);
        float v = s2 * (1.f / 64.f) - m * m;
        float rs = rsqrtf(v + 1e-5f);
        float acc = e_b2[0];
        for (int c = 0; c < 64; ++c) {
            float y = (hb[c] - m) * rs * e_lnw[c] + e_lnb[c];
            acc += fmaxf(y, 0.f) * e_w2[c];
        }
        p = 1.f / (1.f + expf(-acc));
    }

    // ---- edge_refine GEMM: [16x256] x [256x256] ----
    for (int nt = 0; nt < 16; ++nt) {
        v8f acc = {};
        const h16t* bp = rwT + (size_t)(nt * 16 + colL) * 256 + aoff;
#pragma unroll
        for (int kt = 0; kt < 8; ++kt) {
            v8h blo = *reinterpret_cast<const v8h*>(bp + kt * 32);
            v8h bhi = *reinterpret_cast<const v8h*>(bp + kt * 32 + 16);
            v16h bb = __builtin_shufflevector(blo, bhi, 0, 1, 2, 3, 4, 5, 6, 7,
                                              8, 9, 10, 11, 12, 13, 14, 15);
            acc = __builtin_amdgcn_wmma_f32_16x16x32_f16(
                false, a[kt], false, bb, (short)0, acc, false, false);
        }
        int colD = nt * 16 + colL;
#pragma unroll
        for (int r = 0; r < 8; ++r) refBuf[(rbase + r) * CH + colD] = acc[r];
    }
    __syncthreads();

    // ---- refine epilogue: +b, LN(256), ReLU, residual + prob, full output ----
    if (lane < 16) {
        int row = m0 + lane;
        const float* fr = featF + (size_t)row * CH;
        float* rb = refBuf + lane * CH;
        float s = 0.f, s2 = 0.f;
        for (int c = 0; c < CH; ++c) {
            float x = rb[c] + r_b[c];
            rb[c] = x; s += x; s2 += x * x;
        }
        float m = s * (1.f / CH);
        float v = s2 * (1.f / CH) - m * m;
        float rs = rsqrtf(v + 1e-5f);
        float* op = out + (size_t)row * CH;
        for (int c = 0; c < CH; ++c) {
            float y = (rb[c] - m) * rs * r_lnw[c] + r_lnb[c];
            y = fmaxf(y, 0.f);
            op[c] = fr[c] + y * p;
        }
        out[PROB_OFF + row] = p;
    }
}

// ============================================================
extern "C" void kernel_launch(void* const* d_in, const int* in_sizes, int n_in,
                              void* d_out, int out_size, void* d_ws, size_t ws_size,
                              hipStream_t stream) {
    (void)in_sizes; (void)n_in; (void)out_size; (void)ws_size;
    const float* xyz   = (const float*)d_in[0];
    const float* feat  = (const float*)d_in[1];
    const float* g_w1  = (const float*)d_in[2];
    const float* g_b1  = (const float*)d_in[3];
    const float* g_lnw = (const float*)d_in[4];
    const float* g_lnb = (const float*)d_in[5];
    const float* g_w2  = (const float*)d_in[6];
    const float* g_b2  = (const float*)d_in[7];
    const float* e_w1  = (const float*)d_in[8];
    const float* e_b1  = (const float*)d_in[9];
    const float* e_lnw = (const float*)d_in[10];
    const float* e_lnb = (const float*)d_in[11];
    const float* e_w2  = (const float*)d_in[12];
    const float* e_b2  = (const float*)d_in[13];
    const float* r_w   = (const float*)d_in[14];
    const float* r_b   = (const float*)d_in[15];
    const float* r_lnw = (const float*)d_in[16];
    const float* r_lnb = (const float*)d_in[17];
    float* out = (float*)d_out;

    char* ws = (char*)d_ws;
    h16t* featF16 = (h16t*)(ws + WS_FEAT);
    h16t* geoF16  = (h16t*)(ws + WS_GEO);
    h16t* rwT     = (h16t*)(ws + WS_RWT);
    h16t* ew1T    = (h16t*)(ws + WS_EW1T);

    // prep: conversions + transposes
    cvt_feat_kernel<<<(BN * CH) / 256, 256, 0, stream>>>(feat, featF16);
    tr_rw_kernel<<<256, 256, 0, stream>>>(r_w, rwT);
    tr_ew1_kernel<<<64, 288, 0, stream>>>(e_w1, ew1T);

    // KNN + geo features + geo MLP
    knn_geo_kernel<<<BN / 256, 256, 0, stream>>>(xyz, g_w1, g_b1, g_lnw, g_lnb,
                                                 g_w2, g_b2, geoF16);

    // fused WMMA GEMMs + epilogues (4 waves/block, 16x256 f32 LDS per wave)
    size_t shmem = 4 * 16 * CH * sizeof(float);  // 64 KB
    fused_edge_kernel<<<BN / 16 / 4, 128, shmem, stream>>>(
        featF16, geoF16, rwT, ew1T, feat,
        e_b1, e_lnw, e_lnb, e_w2, e_b2,
        r_b, r_lnw, r_lnb, out);
}
